// AffinityHead_3556232921657
// MI455X (gfx1250) — compile-verified, matrix operand order checked by reading, not run on hardware
//
#include <hip/hip_runtime.h>
#include <hip/hip_bf16.h>

// MI455X / gfx1250 implementation. All heavy GEMMs use v_wmma_f32_16x16x32_bf16.
// Workspace (~49 MB): Z0 f32 [768][128][128] | bf16 weight arena | binidx u8 | bias LUT | sigma[16].

typedef __bf16 bf16_t;
typedef __attribute__((ext_vector_type(16))) __bf16 v16bf;
typedef __attribute__((ext_vector_type(8)))  float  v8f;

#define HDIM 128
#define PROT 128
#define NHEAD 4
#define HEADD 32
#define FFD 512
#define NREAL 768
#define APL 48

__device__ __forceinline__ v8f wmma_bf16(v16bf a, v16bf b, v8f c) {
  return __builtin_amdgcn_wmma_f32_16x16x32_bf16(false, a, false, b, (short)0, c, false, false);
}

// A-fragment loader (16x32 bf16 tile). Also used for B by storing B transposed
// as [N][K] (weights pre-transposed; k/v tiles stored accordingly).
// ISA 7.12.2: lanes 0-15 row M=lane, K 0-7 in v0..3, 16-23 in v4..7;
//             lanes 16-31 same rows, K 8-15 / 24-31.
__device__ __forceinline__ v16bf ldfrag(const bf16_t* base, int ld) {
  const int lane = threadIdx.x & 31;
  const bf16_t* p = base + (lane & 15) * ld + ((lane >> 4) << 3);
  v16bf f;
#pragma unroll
  for (int j = 0; j < 8; ++j) f[j] = p[j];
#pragma unroll
  for (int j = 0; j < 8; ++j) f[8 + j] = p[16 + j];
  return f;
}

// ---------------------------------------------------------------- prep kernel
struct PrepArgs {
  const float* pos;
  const float* Wp; const float* bp;
  const float* Wb0; const float* Wb1;
  const float* Wi;
  const float* Wq[2]; const float* Wk[2]; const float* Wv[2]; const float* Wg[2];
  const float* Wo[2]; const float* Wt1[2]; const float* Wt2[2];
  bf16_t* arena; unsigned char* binidx; float* btable; float* sigma;
};

__device__ void convT(const float* in, bf16_t* out, int K, int O, int gid, int stride) {
  // out[o*K + k] = bf16(in[k*O + o])  (store weights as [out][in])
  for (int i = gid; i < K * O; i += stride) {
    int o = i / K, k = i - o * K;
    out[i] = (bf16_t)in[k * O + o];
  }
}

__global__ __launch_bounds__(256) void prep_kernel(PrepArgs A) {
  int gid = blockIdx.x * 256 + threadIdx.x;
  int stride = gridDim.x * 256;
  if (gid < 16) A.sigma[gid] = 0.f;
  // distance bins
  for (int i = gid; i < PROT * PROT; i += stride) {
    int p = i >> 7, q = i & 127;
    float dx = A.pos[p*3+0] - A.pos[q*3+0];
    float dy = A.pos[p*3+1] - A.pos[q*3+1];
    float dz = A.pos[p*3+2] - A.pos[q*3+2];
    float d = sqrtf(fmaxf(dx*dx + dy*dy + dz*dz, 1e-12f));
    int b = (int)(d * (16.f / 30.f));
    b = b < 0 ? 0 : (b > 15 ? 15 : b);
    A.binidx[i] = (unsigned char)b;
  }
  // pair-bias LUT: bias[blk][h][bin] = (Wp[bin,:]+bp) . Wb[:,h]
  for (int i = gid; i < 2 * NHEAD * 16; i += stride) {
    int blk = i >> 6, rem = i & 63;
    int h = rem >> 4, bin = rem & 15;
    const float* Wb = blk ? A.Wb1 : A.Wb0;
    float s = 0.f;
    for (int c = 0; c < HDIM; ++c) s += (A.Wp[bin*HDIM + c] + A.bp[c]) * Wb[c*NHEAD + h];
    A.btable[(blk*NHEAD + h)*16 + bin] = s;
  }
  // weight conversions (transposed bf16)
  convT(A.Wi, A.arena, 128, 128, gid, stride);
  for (int blk = 0; blk < 2; ++blk) {
    bf16_t* bb = A.arena + 16384 + blk * 212992;
    convT(A.Wq[blk],  bb +      0, 128, 128, gid, stride);
    convT(A.Wk[blk],  bb +  16384, 128, 128, gid, stride);
    convT(A.Wv[blk],  bb +  32768, 128, 128, gid, stride);
    convT(A.Wg[blk],  bb +  49152, 128, 128, gid, stride);
    convT(A.Wo[blk],  bb +  65536, 128, 128, gid, stride);
    convT(A.Wt1[blk], bb +  81920, 128, 512, gid, stride);
    convT(A.Wt2[blk], bb + 147456, 512, 128, gid, stride);
  }
}

// ------------------------------------------------------------ interaction MLP
__global__ __launch_bounds__(256) void interact_kernel(
    const float* __restrict__ Xl, const float* __restrict__ Xp,
    const bf16_t* __restrict__ WiT, const float* __restrict__ bi,
    float* __restrict__ Z0) {
  __shared__ bf16_t Abuf[PROT * HDIM];   // [p][k] = Xp[p,k]*Xl[r,k]
  __shared__ float xl[HDIM];
  const int r = blockIdx.x;
  const int tid = threadIdx.x, lane = tid & 31, wid = tid >> 5;
  if (tid < HDIM) xl[tid] = Xl[r * HDIM + tid];
  __syncthreads();
  for (int i = tid; i < PROT * HDIM; i += 256) {
    int p = i >> 7, k = i & 127;
    Abuf[i] = (bf16_t)(xl[k] * Xp[p * HDIM + k]);
  }
  __syncthreads();
  float* out = Z0 + (size_t)r * (PROT * HDIM);
  for (int nt = 0; nt < 8; ++nt) {
    v8f acc = {};
    for (int kk = 0; kk < HDIM; kk += 32) {
      v16bf a = ldfrag(Abuf + (wid*16)*HDIM + kk, HDIM);
      v16bf b = ldfrag(WiT + (nt*16)*HDIM + kk, HDIM);
      acc = wmma_bf16(a, b, acc);
    }
    int col = nt*16 + (lane & 15);
    float bias = bi[col];
    int rbase = wid*16 + ((lane >> 4) << 3);
#pragma unroll
    for (int j = 0; j < 8; ++j) {
      float v = acc[j] + bias;
      out[(rbase + j)*HDIM + col] = v > 0.f ? v : 0.01f * v;   // leaky_relu
    }
  }
}

// -------------------------------------------------------- transformer blocks
struct CArgs {
  const float* Z0;
  const unsigned char* binidx;
  const float* btable;
  float* sigma;
  const float *sigW, *sigb, *gateW, *gateb;
  const float *ln1g[2], *ln1b[2], *ln2g[2], *ln2b[2];
  const float *bq[2], *bk[2], *bv[2], *bg[2], *bo[2], *bt1[2], *bt2[2];
  const bf16_t *WqT[2], *WkT[2], *WvT[2], *WgT[2], *WoT[2], *Wt1T[2], *Wt2T[2];
};

__device__ __forceinline__ void layernorm_to(const bf16_t* Z, bf16_t* Zn,
                                             const float* g, const float* b, int tid) {
  if (tid < PROT) {
    const bf16_t* row = Z + tid * HDIM;
    float mu = 0.f;
    for (int h = 0; h < HDIM; ++h) mu += (float)row[h];
    mu *= (1.f / HDIM);
    float var = 0.f;
    for (int h = 0; h < HDIM; ++h) { float d = (float)row[h] - mu; var += d * d; }
    float rs = rsqrtf(var * (1.f / HDIM) + 1e-5f);
    for (int h = 0; h < HDIM; ++h)
      Zn[tid*HDIM + h] = (bf16_t)(((float)row[h] - mu) * rs * g[h] + b[h]);
  }
}

__global__ __launch_bounds__(256) void blocks_kernel(CArgs A) {
  // 160 KB LDS total: residual tile + overlaid scratch.
  __shared__ bf16_t Zres[PROT * HDIM];                    // 32 KB
  __shared__ __align__(16) unsigned char scr[131072];     // 128 KB
  bf16_t* Zn    = (bf16_t*)scr;                 // [128][128]
  bf16_t* qh    = (bf16_t*)(scr + 32768);       // [128][32]
  bf16_t* khb   = (bf16_t*)(scr + 40960);       // [128][32]
  bf16_t* vTb   = (bf16_t*)(scr + 49152);       // [32][128] (transposed v)
  bf16_t* ghb   = (bf16_t*)(scr + 57344);       // [128][32]
  bf16_t* attnb = (bf16_t*)(scr + 65536);       // [128][128] scores/attn
  bf16_t* Obuf  = (bf16_t*)(scr + 98304);       // [128][128]
  bf16_t* Hid   = (bf16_t*)(scr + 65536);       // [128][256] FFN phase (overlays attnb/Obuf)

  const int tid = threadIdx.x, lane = tid & 31, wid = tid >> 5;
  const int r = blockIdx.x;
  const int lig = r / APL;
  const float* z0 = A.Z0 + (size_t)r * (PROT * HDIM);
  const float scale = 0.17677669529663687f;   // 1/sqrt(32)

  for (int i = tid; i < PROT * HDIM; i += 256) Zres[i] = (bf16_t)z0[i];
  __syncthreads();

  for (int blk = 0; blk < 2; ++blk) {
    layernorm_to(Zres, Zn, A.ln1g[blk], A.ln1b[blk], tid);
    __syncthreads();

    for (int head = 0; head < NHEAD; ++head) {
      const bf16_t* Wproj[4] = { A.WqT[blk], A.WkT[blk], A.WvT[blk], A.WgT[blk] };
      const float*  bproj[4] = { A.bq[blk],  A.bk[blk],  A.bv[blk],  A.bg[blk]  };
#pragma unroll
      for (int which = 0; which < 4; ++which) {
        for (int nt = 0; nt < 2; ++nt) {
          v8f acc = {};
          for (int kk = 0; kk < HDIM; kk += 32) {
            v16bf a = ldfrag(Zn + (wid*16)*HDIM + kk, HDIM);
            v16bf b = ldfrag(Wproj[which] + (head*HEADD + nt*16)*HDIM + kk, HDIM);
            acc = wmma_bf16(a, b, acc);
          }
          int col = nt*16 + (lane & 15);
          float bias = bproj[which][head*HEADD + col];
          int rbase = wid*16 + ((lane >> 4) << 3);
#pragma unroll
          for (int j = 0; j < 8; ++j) {
            float v = acc[j] + bias;
            int row = rbase + j;
            if      (which == 0) qh [row*HEADD + col] = (bf16_t)v;
            else if (which == 1) khb[row*HEADD + col] = (bf16_t)v;
            else if (which == 2) vTb[col*PROT + row]  = (bf16_t)v;     // store v transposed
            else                 ghb[row*HEADD + col] = (bf16_t)(1.f / (1.f + __expf(-v)));
          }
        }
      }
      __syncthreads();

      // scores = q.k^T * scale + pair-bias LUT  -> attnb
      const float* tbl = A.btable + (blk*NHEAD + head)*16;
      for (int nt = 0; nt < 8; ++nt) {
        v8f acc = {};
        v16bf a = ldfrag(qh + (wid*16)*HEADD, HEADD);
        v16bf b = ldfrag(khb + (nt*16)*HEADD, HEADD);
        acc = wmma_bf16(a, b, acc);
        int col = nt*16 + (lane & 15);
        int rbase = wid*16 + ((lane >> 4) << 3);
#pragma unroll
        for (int j = 0; j < 8; ++j) {
          int row = rbase + j;
          attnb[row*PROT + col] = (bf16_t)(acc[j]*scale + tbl[A.binidx[row*PROT + col]]);
        }
      }
      __syncthreads();

      // row softmax in place
      if (tid < PROT) {
        bf16_t* srow = attnb + tid * PROT;
        float m = -3.0e38f;
        for (int q2 = 0; q2 < PROT; ++q2) m = fmaxf(m, (float)srow[q2]);
        float s = 0.f;
        for (int q2 = 0; q2 < PROT; ++q2) { float e = __expf((float)srow[q2] - m); s += e; srow[q2] = (bf16_t)e; }
        float inv = 1.f / s;
        for (int q2 = 0; q2 < PROT; ++q2) srow[q2] = (bf16_t)((float)srow[q2] * inv);
      }
      __syncthreads();

      // o = attn @ v, gated, into Obuf column band of this head
      for (int nt = 0; nt < 2; ++nt) {
        v8f acc = {};
        for (int kk = 0; kk < PROT; kk += 32) {
          v16bf a = ldfrag(attnb + (wid*16)*PROT + kk, PROT);
          v16bf b = ldfrag(vTb + (nt*16)*PROT + kk, PROT);
          acc = wmma_bf16(a, b, acc);
        }
        int col = nt*16 + (lane & 15);
        int rbase = wid*16 + ((lane >> 4) << 3);
#pragma unroll
        for (int j = 0; j < 8; ++j) {
          int row = rbase + j;
          float g = (float)ghb[row*HEADD + col];
          Obuf[row*HDIM + head*HEADD + col] = (bf16_t)(acc[j] * g);
        }
      }
      __syncthreads();
    } // head

    // Zres += Obuf @ Wo + bo
    for (int nt = 0; nt < 8; ++nt) {
      v8f acc = {};
      for (int kk = 0; kk < HDIM; kk += 32) {
        v16bf a = ldfrag(Obuf + (wid*16)*HDIM + kk, HDIM);
        v16bf b = ldfrag(A.WoT[blk] + (nt*16)*HDIM + kk, HDIM);
        acc = wmma_bf16(a, b, acc);
      }
      int col = nt*16 + (lane & 15);
      float bias = A.bo[blk][col];
      int rbase = wid*16 + ((lane >> 4) << 3);
#pragma unroll
      for (int j = 0; j < 8; ++j) {
        int idx = (rbase + j)*HDIM + col;
        Zres[idx] = (bf16_t)((float)Zres[idx] + acc[j] + bias);
      }
    }
    __syncthreads();

    layernorm_to(Zres, Zn, A.ln2g[blk], A.ln2b[blk], tid);
    __syncthreads();

    // FFN in two 256-wide K chunks (Hid overlays attention scratch)
    for (int chunk = 0; chunk < 2; ++chunk) {
      for (int nt = 0; nt < 16; ++nt) {
        v8f acc = {};
        for (int kk = 0; kk < HDIM; kk += 32) {
          v16bf a = ldfrag(Zn + (wid*16)*HDIM + kk, HDIM);
          v16bf b = ldfrag(A.Wt1T[blk] + (chunk*256 + nt*16)*HDIM + kk, HDIM);
          acc = wmma_bf16(a, b, acc);
        }
        int col = nt*16 + (lane & 15);
        float bias = A.bt1[blk][chunk*256 + col];
        int rbase = wid*16 + ((lane >> 4) << 3);
#pragma unroll
        for (int j = 0; j < 8; ++j)
          Hid[(rbase + j)*256 + col] = (bf16_t)fmaxf(acc[j] + bias, 0.f);
      }
      __syncthreads();
      for (int nt = 0; nt < 8; ++nt) {
        v8f acc = {};
        for (int kk = 0; kk < 256; kk += 32) {
          v16bf a = ldfrag(Hid + (wid*16)*256 + kk, 256);
          v16bf b = ldfrag(A.Wt2T[blk] + (nt*16)*FFD + chunk*256 + kk, FFD);
          acc = wmma_bf16(a, b, acc);
        }
        int col = nt*16 + (lane & 15);
        float bias = (chunk == 0) ? A.bt2[blk][col] : 0.f;
        int rbase = wid*16 + ((lane >> 4) << 3);
#pragma unroll
        for (int j = 0; j < 8; ++j) {
          int idx = (rbase + j)*HDIM + col;
          Zres[idx] = (bf16_t)((float)Zres[idx] + acc[j] + bias);
        }
      }
      __syncthreads();
    }
  } // blk

  // gated sigma head on Zc_out = Z0 + Z, reduced over protein rows -> sigma[lig]
  float* red = (float*)scr;
  float part = 0.f;
  if (tid < PROT) {
    const float* z0r = z0 + tid * HDIM;
    float ds = A.sigb[0], dg = A.gateb[0];
    for (int h = 0; h < HDIM; ++h) {
      float zc = z0r[h] + (float)Zres[tid*HDIM + h];
      ds += zc * A.sigW[h];
      dg += zc * A.gateW[h];
    }
    part = ds * (1.f / (1.f + __expf(-dg)));
  }
  __syncthreads();
  if (tid < PROT) red[tid] = part;
  __syncthreads();
  if (tid == 0) {
    float s = 0.f;
    for (int i = 0; i < PROT; ++i) s += red[i];
    atomicAdd(A.sigma + lig, s);
  }
}

// ---------------------------------------------------------------- final head
__global__ __launch_bounds__(256) void final_kernel(
    const float* Zp, const float* Zl,
    const float* muW1, const float* mub1, const float* muW2, const float* mub2,
    const float* stdW1, const float* stdb1, const float* stdW2, const float* stdb2,
    const float* sbW1, const float* sbb1, const float* sbW2, const float* sbb2,
    const float* sigma, float* out) {
  __shared__ float t[HDIM];
  __shared__ float sc[2];
  int tid = threadIdx.x;
  for (int pass = 0; pass < 2; ++pass) {
    const float* W1 = pass ? stdW1 : muW1; const float* b1 = pass ? stdb1 : mub1;
    const float* W2 = pass ? stdW2 : muW2; const float* b2 = pass ? stdb2 : mub2;
    if (tid < HDIM) {
      float s = b1[tid];
      for (int k = 0; k < HDIM; ++k) s += Zp[k] * W1[k*HDIM + tid];
      s = s > 0.f ? s : 0.01f * s;
      t[tid] = s * W2[tid];
    }
    __syncthreads();
    if (tid == 0) {
      float s = b2[0];
      for (int i = 0; i < HDIM; ++i) s += t[i];
      sc[pass] = 1.f / (1.f + __expf(-s));
    }
    __syncthreads();
  }
  float mu = sc[0] * -15.f, sd = sc[1] * 5.f;
  for (int b = 0; b < 16; ++b) {
    if (tid < HDIM) {
      float s = sbb1[tid];
      for (int k = 0; k < HDIM; ++k) s += Zp[k] * sbW1[k*HDIM + tid];
      for (int k = 0; k < HDIM; ++k) s += Zl[b*HDIM + k] * sbW1[(HDIM + k)*HDIM + tid];
      s = s > 0.f ? s : 0.01f * s;
      t[tid] = s * sbW2[tid];
    }
    __syncthreads();
    if (tid == 0) {
      float s = sbb2[0];
      for (int i = 0; i < HDIM; ++i) s += t[i];
      out[b] = (sigma[b] + s) * sd + mu;
    }
    __syncthreads();
  }
}

// --------------------------------------------------------------------- launch
extern "C" void kernel_launch(void* const* d_in, const int* in_sizes, int n_in,
                              void* d_out, int out_size, void* d_ws, size_t ws_size,
                              hipStream_t stream) {
  (void)in_sizes; (void)n_in; (void)out_size; (void)ws_size;
  auto F = [&](int i) { return (const float*)d_in[i]; };
  // setup_inputs() insertion order:
  const float* Xp  = F(0);  // X_protein [128,128]
  const float* pos = F(1);  // pos_protein [128,3]
  const float* Zp  = F(2);  // Z_protein [128]
  const float* Xl  = F(3);  // X_ligand [768,128]
  const float* Zl  = F(4);  // Z_ligand [16,128]
  // d_in[5] = ligand_batch (structure is static: 48 atoms/ligand) -- unused
  const float* Wi  = F(6);  const float* bi = F(7);
  const float* Wp  = F(8);  const float* bp = F(9);
  // blocks: 19 params each starting at 10 / 29
  // +0 ln1_g +1 ln1_b +2 Wq +3 bq +4 Wk +5 bk +6 Wv +7 bv +8 Wg +9 bg
  // +10 Wb +11 Wo +12 bo +13 ln2_g +14 ln2_b +15 Wt1 +16 bt1 +17 Wt2 +18 bt2

  unsigned char* ws = (unsigned char*)d_ws;
  float* Z0            = (float*)ws;                                  // 50,331,648 B
  bf16_t* arena        = (bf16_t*)(ws + 50331648);                    //    884,736 B
  unsigned char* binix = ws + 51216384;                               //     16,384 B
  float* btable        = (float*)(ws + 51232768);                     //        512 B
  float* sigma         = (float*)(ws + 51233280);                     //         64 B

  PrepArgs PA;
  PA.pos = pos; PA.Wp = Wp; PA.bp = bp;
  PA.Wb0 = F(10 + 10); PA.Wb1 = F(29 + 10);
  PA.Wi = Wi;
  for (int blk = 0; blk < 2; ++blk) {
    int b0 = 10 + blk * 19;
    PA.Wq[blk]  = F(b0 + 2);  PA.Wk[blk]  = F(b0 + 4);
    PA.Wv[blk]  = F(b0 + 6);  PA.Wg[blk]  = F(b0 + 8);
    PA.Wo[blk]  = F(b0 + 11); PA.Wt1[blk] = F(b0 + 15); PA.Wt2[blk] = F(b0 + 17);
  }
  PA.arena = arena; PA.binidx = binix; PA.btable = btable; PA.sigma = sigma;
  prep_kernel<<<64, 256, 0, stream>>>(PA);

  interact_kernel<<<NREAL, 256, 0, stream>>>(Xl, Xp, arena, bi, Z0);

  CArgs CA;
  CA.Z0 = Z0; CA.binidx = binix; CA.btable = btable; CA.sigma = sigma;
  CA.sigW = F(60); CA.sigb = F(61); CA.gateW = F(62); CA.gateb = F(63);
  for (int blk = 0; blk < 2; ++blk) {
    int b0 = 10 + blk * 19;
    CA.ln1g[blk] = F(b0 + 0);  CA.ln1b[blk] = F(b0 + 1);
    CA.bq[blk] = F(b0 + 3);  CA.bk[blk] = F(b0 + 5);
    CA.bv[blk] = F(b0 + 7);  CA.bg[blk] = F(b0 + 9);
    CA.bo[blk] = F(b0 + 12);
    CA.ln2g[blk] = F(b0 + 13); CA.ln2b[blk] = F(b0 + 14);
    CA.bt1[blk] = F(b0 + 16); CA.bt2[blk] = F(b0 + 18);
    bf16_t* bb = arena + 16384 + blk * 212992;
    CA.WqT[blk]  = bb;          CA.WkT[blk] = bb + 16384;
    CA.WvT[blk]  = bb + 32768;  CA.WgT[blk] = bb + 49152;
    CA.WoT[blk]  = bb + 65536;  CA.Wt1T[blk] = bb + 81920;
    CA.Wt2T[blk] = bb + 147456;
  }
  blocks_kernel<<<NREAL, 256, 0, stream>>>(CA);

  final_kernel<<<1, 256, 0, stream>>>(
      Zp, Zl,
      F(48), F(49), F(50), F(51),       // mu
      F(52), F(53), F(54), F(55),       // std
      F(56), F(57), F(58), F(59),       // sigma bias MLP
      sigma, (float*)d_out);
}
